// CAM_Module_90211493085322
// MI455X (gfx1250) — matrix operand order repaired and checked by reading, not compiled, not run on hardware
//
#include <hip/hip_runtime.h>

typedef __attribute__((ext_vector_type(16))) __bf16 v16bf;
typedef __attribute__((ext_vector_type(8)))  __bf16 v8bf;
typedef __attribute__((ext_vector_type(8)))  float  v8f;

namespace {
constexpr int kB = 8;
constexpr int kC = 512;
constexpr int kN = 4096;   // H*W
}

union BfVec  { uint4 q[2]; v16bf v; };
union BfVec8 { uint4 q;    v8bf  v; };

// ---------------------------------------------------------------------------
// Kernel 0: bulk f32 -> bf16 conversion (x2 -> qbf). 8 elements / thread.
// ---------------------------------------------------------------------------
__global__ void __launch_bounds__(256)
cam_cvt_kernel(const float* __restrict__ in, __bf16* __restrict__ out) {
  const size_t t = (size_t)blockIdx.x * 256 + threadIdx.x;
  const float4* p = (const float4*)(in + t * 8);
  float4 f0 = p[0];
  float4 f1 = p[1];
  BfVec8 o;
  o.v[0] = (__bf16)f0.x; o.v[1] = (__bf16)f0.y;
  o.v[2] = (__bf16)f0.z; o.v[3] = (__bf16)f0.w;
  o.v[4] = (__bf16)f1.x; o.v[5] = (__bf16)f1.y;
  o.v[6] = (__bf16)f1.z; o.v[7] = (__bf16)f1.w;
  *(uint4*)(out + t * 8) = o.q;
}

// ---------------------------------------------------------------------------
// Kernel 1: energy[b] = Q * Q^T with Q = qbf[b] ([C][N] bf16).
// WG = 64(m) x 128(n) tile, 8 waves; each wave owns a 32x32 block:
// 2 A operands x 2 B operands -> 4 WMMAs per K-step, pure bf16 b128 loads.
// A lane layout: K-chunks [k+hi*8,+8) and [k+hi*8+16,+8).
// B lane layout: K-chunks [k+hi*8,+8) and [k+hi*8+16,+8) of column n == row n of Q.
// ---------------------------------------------------------------------------
__global__ void __launch_bounds__(256)
cam_energy_kernel(const __bf16* __restrict__ qbf, float* __restrict__ energy) {
  const int b  = blockIdx.z;
  const int m0 = blockIdx.y * 64;
  const int n0 = blockIdx.x * 128;
  const int w    = (int)threadIdx.x >> 5;
  const int lane = (int)threadIdx.x & 31;
  const int wm = (w & 1) * 32;      // row block within WG tile
  const int wn = (w >> 1) * 32;     // col block within WG tile
  const int lr = lane & 15;
  const int hi = lane >> 4;

  const __bf16* Q = qbf + (size_t)b * kC * kN;
  const __bf16* ra0 = Q + (size_t)(m0 + wm + lr) * kN;        // A rows
  const __bf16* ra1 = ra0 + (size_t)16 * kN;
  const __bf16* rb0 = Q + (size_t)(n0 + wn + lr) * kN;        // B "rows"
  const __bf16* rb1 = rb0 + (size_t)16 * kN;

  v8f acc00 = {}, acc01 = {}, acc10 = {}, acc11 = {};

  for (int kk = 0; kk < kN; kk += 32) {
    const int ka = kk + hi * 8;     // A: chunks at ka and ka+16 (elements)
    const int kb = kk + hi * 8;     // B: chunks at kb and kb+16 (elements)

    BfVec a0, a1, b0, b1;
    {
      const uint4* p = (const uint4*)(ra0 + ka);
      a0.q[0] = p[0]; a0.q[1] = p[2];        // +16 bf16 == +2 uint4
    }
    {
      const uint4* p = (const uint4*)(ra1 + ka);
      a1.q[0] = p[0]; a1.q[1] = p[2];
    }
    {
      const uint4* p = (const uint4*)(rb0 + kb);
      b0.q[0] = p[0]; b0.q[1] = p[2];
    }
    {
      const uint4* p = (const uint4*)(rb1 + kb);
      b1.q[0] = p[0]; b1.q[1] = p[2];
    }
    if (kk + 32 < kN) {   // gfx1250 global_prefetch_b8: next K chunk
      __builtin_prefetch(ra0 + ka + 32, 0, 1);
      __builtin_prefetch(ra1 + ka + 32, 0, 1);
      __builtin_prefetch(rb0 + kb + 32, 0, 1);
      __builtin_prefetch(rb1 + kb + 32, 0, 1);
    }
    acc00 = __builtin_amdgcn_wmma_f32_16x16x32_bf16(false, a0.v, false, b0.v,
                                                    (short)0, acc00, false, false);
    acc01 = __builtin_amdgcn_wmma_f32_16x16x32_bf16(false, a0.v, false, b1.v,
                                                    (short)0, acc01, false, false);
    acc10 = __builtin_amdgcn_wmma_f32_16x16x32_bf16(false, a1.v, false, b0.v,
                                                    (short)0, acc10, false, false);
    acc11 = __builtin_amdgcn_wmma_f32_16x16x32_bf16(false, a1.v, false, b1.v,
                                                    (short)0, acc11, false, false);
  }

  // D layout: VGPR r -> (m = r + 8*hi, n = lane&15)
  float* E = energy + ((size_t)b * kC + (m0 + wm)) * kC + n0 + wn;
  const int mo = hi * 8;
  #pragma unroll
  for (int r = 0; r < 8; ++r) {
    E[(size_t)(mo + r) * kC + lr]           = acc00[r];
    E[(size_t)(mo + r) * kC + 16 + lr]      = acc01[r];
    E[(size_t)(mo + r + 16) * kC + lr]      = acc10[r];
    E[(size_t)(mo + r + 16) * kC + 16 + lr] = acc11[r];
  }
}

// ---------------------------------------------------------------------------
// Kernel 2: row softmax over 512 logits (f32 in, bf16 out). One wave32 per
// row; 16 contiguous elements per lane; shfl_xor reductions (width 32).
// ---------------------------------------------------------------------------
__global__ void __launch_bounds__(256)
cam_softmax_kernel(const float* __restrict__ energy, __bf16* __restrict__ att) {
  const int w    = (int)threadIdx.x >> 5;
  const int lane = (int)threadIdx.x & 31;
  const size_t row = (size_t)blockIdx.x * 8 + w;   // rows of [B*C][C]
  const float* e = energy + row * kC + lane * 16;

  float v[16];
  const float4* p = (const float4*)e;
  #pragma unroll
  for (int i = 0; i < 4; ++i) {
    float4 f = p[i];
    v[4*i+0] = f.x; v[4*i+1] = f.y; v[4*i+2] = f.z; v[4*i+3] = f.w;
  }
  float m = v[0];
  #pragma unroll
  for (int i = 1; i < 16; ++i) m = fmaxf(m, v[i]);
  #pragma unroll
  for (int off = 16; off >= 1; off >>= 1) m = fmaxf(m, __shfl_xor(m, off, 32));

  float s = 0.0f;
  #pragma unroll
  for (int i = 0; i < 16; ++i) { v[i] = __expf(v[i] - m); s += v[i]; }
  #pragma unroll
  for (int off = 16; off >= 1; off >>= 1) s += __shfl_xor(s, off, 32);
  const float inv = 1.0f / s;

  BfVec o;
  #pragma unroll
  for (int i = 0; i < 16; ++i) o.v[i] = (__bf16)(v[i] * inv);
  uint4* po = (uint4*)(att + row * kC + lane * 16);
  po[0] = o.q[0];
  po[1] = o.q[1];
}

// ---------------------------------------------------------------------------
// Kernel 3: out[b] = gamma * (att[b] * V) + x[b], V = x[b] as [C][N] f32.
// WG = 64(m) x 128(n) tile, 8 waves, each a 32x32 block (4 WMMAs/K-step).
// V chunk 32x128 is staged to LDS transposed as bf16 (80B rows, 16B aligned);
// B operands come back via ds_load_b128 pairs.
// ---------------------------------------------------------------------------
__global__ void __launch_bounds__(256)
cam_av_kernel(const __bf16* __restrict__ att, const float* __restrict__ x,
              const float* __restrict__ gamma, float* __restrict__ out) {
  constexpr int VSTR = 40;                        // bf16 stride = 80 B
  __shared__ __align__(16) __bf16 vT[128][VSTR];  // [n_local][k_local]

  const int b  = blockIdx.z;
  const int m0 = blockIdx.y * 64;    // channel rows
  const int n0 = blockIdx.x * 128;   // pixel cols
  const int t    = (int)threadIdx.x;
  const int w    = t >> 5;
  const int lane = t & 31;
  const int wm = (w & 1) * 32;
  const int wn = (w >> 1) * 32;
  const int lr = lane & 15;
  const int hi = lane >> 4;

  const float g = gamma[0];
  const __bf16* Ar0 = att + ((size_t)b * kC + (m0 + wm + lr)) * kC;
  const __bf16* Ar1 = Ar0 + (size_t)16 * kC;
  const float* Vb = x + (size_t)b * kC * kN;

  const int ks = t >> 3;            // 0..31  : k within chunk (staging)
  const int nb = (t & 7) * 16;      // 0..112 : n offset within tile (staging)

  v8f acc00 = {}, acc01 = {}, acc10 = {}, acc11 = {};

  for (int k0 = 0; k0 < kC; k0 += 32) {
    __syncthreads();
    {  // stage V[k0..k0+31][n0..n0+127] -> vT[n][k] bf16 (transpose)
      const float4* p = (const float4*)(Vb + (size_t)(k0 + ks) * kN + n0 + nb);
      #pragma unroll
      for (int j = 0; j < 4; ++j) {
        float4 f = p[j];
        vT[nb + 4*j + 0][ks] = (__bf16)f.x;
        vT[nb + 4*j + 1][ks] = (__bf16)f.y;
        vT[nb + 4*j + 2][ks] = (__bf16)f.z;
        vT[nb + 4*j + 3][ks] = (__bf16)f.w;
      }
    }
    __syncthreads();

    BfVec a0, a1, b0, b1;
    {  // A: bf16 attention rows, chunks at k0+hi*8 and +16
      const uint4* p = (const uint4*)(Ar0 + k0 + hi * 8);
      a0.q[0] = p[0]; a0.q[1] = p[2];
      const uint4* q = (const uint4*)(Ar1 + k0 + hi * 8);
      a1.q[0] = q[0]; a1.q[1] = q[2];
    }
    {  // B: k chunks at hi*8 and hi*8+16 of column n from LDS (ds_load_b128 x2 each)
      const uint4* p = (const uint4*)(&vT[wn + lr][0]);
      b0.q[0] = p[hi]; b0.q[1] = p[hi + 2];
      const uint4* q = (const uint4*)(&vT[wn + 16 + lr][0]);
      b1.q[0] = q[hi]; b1.q[1] = q[hi + 2];
    }
    acc00 = __builtin_amdgcn_wmma_f32_16x16x32_bf16(false, a0.v, false, b0.v,
                                                    (short)0, acc00, false, false);
    acc01 = __builtin_amdgcn_wmma_f32_16x16x32_bf16(false, a0.v, false, b1.v,
                                                    (short)0, acc01, false, false);
    acc10 = __builtin_amdgcn_wmma_f32_16x16x32_bf16(false, a1.v, false, b0.v,
                                                    (short)0, acc10, false, false);
    acc11 = __builtin_amdgcn_wmma_f32_16x16x32_bf16(false, a1.v, false, b1.v,
                                                    (short)0, acc11, false, false);
  }

  const int mo = hi * 8;
  float* Ob       = out + ((size_t)b * kC + (m0 + wm)) * kN + n0 + wn;
  const float* Xb = x   + ((size_t)b * kC + (m0 + wm)) * kN + n0 + wn;
  #pragma unroll
  for (int r = 0; r < 8; ++r) {
    size_t i00 = (size_t)(mo + r) * kN + lr;
    size_t i01 = i00 + 16;
    size_t i10 = (size_t)(mo + r + 16) * kN + lr;
    size_t i11 = i10 + 16;
    Ob[i00] = fmaf(g, acc00[r], Xb[i00]);
    Ob[i01] = fmaf(g, acc01[r], Xb[i01]);
    Ob[i10] = fmaf(g, acc10[r], Xb[i10]);
    Ob[i11] = fmaf(g, acc11[r], Xb[i11]);
  }
}

// ---------------------------------------------------------------------------
extern "C" void kernel_launch(void* const* d_in, const int* in_sizes, int n_in,
                              void* d_out, int out_size, void* d_ws, size_t ws_size,
                              hipStream_t stream) {
  (void)in_sizes; (void)n_in; (void)out_size; (void)ws_size;
  const float* x     = (const float*)d_in[0];
  const float* x2    = (const float*)d_in[1];
  const float* gamma = (const float*)d_in[2];
  float* out = (float*)d_out;

  char* ws = (char*)d_ws;
  constexpr size_t kQBytes = (size_t)kB * kC * kN * 2;       // 32 MB bf16 Q
  constexpr size_t kEBytes = (size_t)kB * kC * kC * 4;       // 8 MB f32 energy
  __bf16* qbf    = (__bf16*)ws;
  float*  energy = (float*)(ws + kQBytes);
  __bf16* att    = (__bf16*)(ws + kQBytes + kEBytes);        // 4 MB bf16

  const int cvt_blocks = (kB * kC * kN) / (256 * 8);         // 8 elems/thread
  cam_cvt_kernel<<<dim3(cvt_blocks), 256, 0, stream>>>(x2, qbf);
  cam_energy_kernel<<<dim3(kC / 128, kC / 64, kB), 256, 0, stream>>>(qbf, energy);
  cam_softmax_kernel<<<dim3((kB * kC) / 8), 256, 0, stream>>>(energy, att);
  cam_av_kernel<<<dim3(kN / 128, kC / 64, kB), 256, 0, stream>>>(att, x, gamma, out);
}
